// SMHSA_71829033059026
// MI455X (gfx1250) — matrix operand order repaired and checked by reading, not compiled, not run on hardware
//
#include <hip/hip_runtime.h>
#include <cstdint>
#include <cstddef>

#define B_   32
#define CIN  64
#define T_   12
#define V_   256
#define COUT 32
#define BT   (B_ * T_)

typedef __attribute__((ext_vector_type(16))) _Float16 v16h;
typedef __attribute__((ext_vector_type(8)))  _Float16 v8h;
typedef __attribute__((ext_vector_type(8)))  float    v8f;
typedef __attribute__((ext_vector_type(4)))  unsigned int v4u;
typedef __attribute__((ext_vector_type(8)))  int          v8i_;
typedef __attribute__((ext_vector_type(4)))  int          v4i_;

// ---- WMMA fragment packing per CDNA5 ISA 7.12.2 (wave32) -------------------
// A (16x32 f16): lane = g*16+M; halves 0..7  = K(8g+0..7), halves 8..15 = K(16+8g+0..7)
static __device__ __forceinline__ v16h pack_a(const _Float16* row, int k0, int g) {
  v8h a0 = *(const v8h*)(row + k0 + 8 * g);
  v8h a1 = *(const v8h*)(row + k0 + 16 + 8 * g);
  v16h a;
#pragma unroll
  for (int h = 0; h < 8; ++h) { a[h] = a0[h]; a[h + 8] = a1[h]; }
  return a;
}
// B (32x16 f16): lane = g*16+N; halves 0..7 = K(8g+0..7), 8..15 = K(16+8g+0..7) of column N -> read Bt[n][k] rows
static __device__ __forceinline__ v16h pack_b(const _Float16* btrow, int k0, int g) {
  v8h b0 = *(const v8h*)(btrow + k0 + 8 * g);
  v8h b1 = *(const v8h*)(btrow + k0 + 16 + 8 * g);
  v16h b;
#pragma unroll
  for (int h = 0; h < 8; ++h) { b[h] = b0[h]; b[h + 8] = b1[h]; }
  return b;
}

// ---- Tensor Data Mover: 2D tile global->LDS with hardware row padding ------
// Tile: tile_dim1 rows x tile_dim0 f16 elems, tensor row stride = stride0 elems.
// Pad 4 dwords after every 128 dwords (512B row) => LDS row stride 264 halves.
static __device__ __forceinline__ void tdm_load_tile_f16(
    const _Float16* gsrc, void* lds_dst) {
  const unsigned long long ga = (unsigned long long)(uintptr_t)gsrc;
  v4u g0;
  g0[0] = 1u;                                          // count=1, user mode
  g0[1] = (unsigned int)(uintptr_t)lds_dst;            // LDS byte address
  g0[2] = (unsigned int)ga;                            // global_addr[31:0]
  g0[3] = (unsigned int)((ga >> 32) & 0x01FFFFFFu)     // global_addr[56:32]
        | 0x80000000u;                                 // type=2 ("image")
  v8i_ g1;
  g1[0] = (1 << 16)      // data_size = 1 (2 bytes)
        | (1 << 20)      // pad_enable
        | (6 << 22)      // pad_interval: 128 dwords (one 512B row)
        | (3 << 25);     // pad_amount: 4 dwords (16B)
  g1[1] = (int)(256u << 16);   // tensor_dim0[15:0]=256 in bits[63:48]
  g1[2] = (int)(32u << 16);    // tensor_dim1[15:0]=32  in bits[95:80]
  g1[3] = (int)(256u << 16);   // tile_dim0=256         in bits[127:112]
  g1[4] = 32;                  // tile_dim1=32          in bits[143:128]
  g1[5] = (int)(T_ * V_);      // tensor_dim0_stride = 3072 elems
  g1[6] = 0;
  g1[7] = 0;
  v4i_ z4 = {0, 0, 0, 0};                 // 2-D tensor: groups 2/3 unused
  v8i_ z8 = {0, 0, 0, 0, 0, 0, 0, 0};     // trailing words (6-arg builtin form)
  __builtin_amdgcn_tensor_load_to_lds(g0, g1, z4, z4, z8, 0);
}

// ---------------------------------------------------------------------------
// Kernel 1: Q = xp @ Wq^T + bq, K = xp @ Wk^T + bk  (per (b,t) tile of 256x64)
// ---------------------------------------------------------------------------
__global__ __launch_bounds__(256) void k_proj(
    const float* __restrict__ x,
    const float* __restrict__ Wq, const float* __restrict__ bq,
    const float* __restrict__ Wk, const float* __restrict__ bk,
    _Float16* __restrict__ Qh, _Float16* __restrict__ Kh)
{
  __shared__ _Float16 xp[V_][72];        // [v][c], padded stride (16B aligned)
  __shared__ _Float16 wl[2][CIN][CIN];   // W row-major = Bt[n=co][k=c]
  const int tid = threadIdx.x;
  const int bt = blockIdx.x, b = bt / T_, t = bt % T_;

  const float* xbt = x + ((size_t)b * CIN * T_ + t) * V_;   // x[b, c, t, v]
#pragma unroll 4
  for (int c = 0; c < CIN; ++c)
    xp[tid][c] = (_Float16)xbt[(size_t)c * (T_ * V_) + tid];
  for (int i = tid; i < CIN * CIN; i += 256) {
    wl[0][i >> 6][i & 63] = (_Float16)Wq[i];
    wl[1][i >> 6][i & 63] = (_Float16)Wk[i];
  }
  __syncthreads();

  const int wave = tid >> 5, lane = tid & 31, g = lane >> 4, ln = lane & 15;
  for (int i = 0; i < 16; ++i) {                 // 128 tiles / 8 waves
    const int tile = wave * 16 + i;
    const int sel = tile >> 6;                   // 0 = Q, 1 = K
    const int m0 = ((tile >> 2) & 15) * 16;      // v tile
    const int n0 = (tile & 3) * 16;              // co tile
    v8f acc = {};
#pragma unroll
    for (int k0 = 0; k0 < CIN; k0 += 32) {
      v16h a = pack_a(&xp[m0 + ln][0], k0, g);
      v16h bf = pack_b(&wl[sel][n0 + ln][0], k0, g);
      acc = __builtin_amdgcn_wmma_f32_16x16x32_f16(false, a, false, bf, (short)0, acc, false, false);
    }
    const float bb = (sel ? bk : bq)[n0 + ln];
    _Float16* dst = (sel ? Kh : Qh) + ((size_t)bt * V_ + m0) * CIN + n0 + ln;
#pragma unroll
    for (int r = 0; r < 8; ++r)
      dst[(size_t)(r + 8 * g) * CIN] = (_Float16)(acc[r] + bb);
  }
}

// ---------------------------------------------------------------------------
// Kernel 2: G[98304,32] = x.reshape(-1,64) @ weight  (flat buffer == x_first)
// f16 output: halves HBM traffic and enables raw TDM staging in k_attn.
// ---------------------------------------------------------------------------
__global__ __launch_bounds__(256) void k_gcn(
    const float* __restrict__ x, const float* __restrict__ w, _Float16* __restrict__ Gh)
{
  __shared__ _Float16 wt[COUT][CIN];   // wt[co][j] = weight[j][co]  (Bt)
  const int tid = threadIdx.x;
  for (int i = tid; i < CIN * COUT; i += 256)
    wt[i & 31][i >> 5] = (_Float16)w[i];
  __syncthreads();

  const int wave = tid >> 5, lane = tid & 31, g = lane >> 4, ln = lane & 15;
  const int r0 = blockIdx.x << 10;               // 1024 rows / block, 96 blocks
  for (int i = 0; i < 16; ++i) {                 // 64 Mtiles x 2 Ntiles / 8 waves
    const int tile = wave * 16 + i;
    const int m0 = r0 + (tile >> 1) * 16;
    const int n0 = (tile & 1) * 16;
    v8f acc = {};
#pragma unroll
    for (int k0 = 0; k0 < CIN; k0 += 32) {
      const float* ap = x + (size_t)(m0 + ln) * CIN + k0 + 8 * g;
      v16h a;
#pragma unroll
      for (int h = 0; h < 8; ++h) { a[h] = (_Float16)ap[h]; a[h + 8] = (_Float16)ap[h + 16]; }
      v16h bf = pack_b(&wt[n0 + ln][0], k0, g);
      acc = __builtin_amdgcn_wmma_f32_16x16x32_f16(false, a, false, bf, (short)0, acc, false, false);
    }
#pragma unroll
    for (int r = 0; r < 8; ++r)
      Gh[(size_t)(m0 + r + 8 * g) * COUT + n0 + ln] = (_Float16)acc[r];
  }
}

// ---------------------------------------------------------------------------
// Kernel 3 (fused, per (b,t)): S = QK^T/4 -> softmax -> out = x_first @ P + bias
// Attention matrix LDS-resident; x_first tile arrives via async TDM overlapped
// with the scores GEMM + softmax.
// ---------------------------------------------------------------------------
__global__ __launch_bounds__(256) void k_attn(
    const _Float16* __restrict__ Qh, const _Float16* __restrict__ Kh,
    const _Float16* __restrict__ Gh, const float* __restrict__ bias,
    float* __restrict__ out)
{
  __shared__ _Float16 S[V_][258];      // 258-half stride: 129 dwords -> conflict-free columns
  __shared__ _Float16 xf[COUT][264];   // x_first tile; 264 = 256 + TDM pad (4 dw / row)
  const int tid = threadIdx.x;
  const int bt = blockIdx.x, b = bt / T_, t = bt % T_;
  const int wave = tid >> 5, lane = tid & 31, g = lane >> 4, ln = lane & 15;
  const _Float16* Qbt = Qh + (size_t)bt * V_ * CIN;
  const _Float16* Kbt = Kh + (size_t)bt * V_ * CIN;

  // Kick off async TDM copy of x_first[b,:,t,:] -> LDS (one wave issues; EXEC-independent)
  if (wave == 0)
    tdm_load_tile_f16(Gh + (((size_t)b * COUT) * T_ + t) * V_, &xf[0][0]);

  // Pull Q/K tiles toward the WGP: each thread prefetches one 128B row
  __builtin_prefetch(Qbt + (size_t)tid * CIN, 0, 0);
  __builtin_prefetch(Kbt + (size_t)tid * CIN, 0, 0);

  // ---- scores: 256 16x16 tiles, 32 per wave ----
  for (int i = 0; i < 32; ++i) {
    const int tile = wave * 32 + i;
    const int m0 = (tile >> 4) * 16, n0 = (tile & 15) * 16;
    v8f acc = {};
#pragma unroll
    for (int k0 = 0; k0 < CIN; k0 += 32) {
      v16h a  = pack_a(Qbt + (size_t)(m0 + ln) * CIN, k0, g);  // A = Q
      v16h bf = pack_b(Kbt + (size_t)(n0 + ln) * CIN, k0, g);  // Bt = K rows
      acc = __builtin_amdgcn_wmma_f32_16x16x32_f16(false, a, false, bf, (short)0, acc, false, false);
    }
#pragma unroll
    for (int r = 0; r < 8; ++r)
      S[m0 + r + 8 * g][n0 + ln] = (_Float16)(acc[r] * 0.25f);  // 1/sqrt(d_k)
  }
  __syncthreads();

  // ---- softmax: one row per thread, f32 math, f16 storage ----
  {
    _Float16* row = S[tid];
    float mx = -3.0e38f;
    for (int j = 0; j < V_; ++j) mx = fmaxf(mx, (float)row[j]);
    float sum = 0.f;
    for (int j = 0; j < V_; ++j) { float e = __expf((float)row[j] - mx); sum += e; row[j] = (_Float16)e; }
    const float inv = 1.0f / sum;
    for (int j = 0; j < V_; ++j) row[j] = (_Float16)((float)row[j] * inv);
  }

  // TDM must have landed before anyone reads xf
  if (wave == 0) __builtin_amdgcn_s_wait_tensorcnt(0);
  __syncthreads();

  // ---- aggregate: [32 x 256] = xf @ P ; 32 tiles, 4 per wave ----
  for (int i = 0; i < 4; ++i) {
    const int tile = wave * 4 + i;
    const int m0 = (tile >> 4) * 16, n0 = (tile & 15) * 16;
    v8f acc = {};
    for (int k0 = 0; k0 < V_; k0 += 32) {
      v16h a = pack_a(&xf[m0 + ln][0], k0, g);
      v16h bf;
#pragma unroll
      for (int h = 0; h < 16; ++h) bf[h] = S[k0 + 8 * g + (h < 8 ? h : h + 8)][n0 + ln];  // B[k][n] column read
      acc = __builtin_amdgcn_wmma_f32_16x16x32_f16(false, a, false, bf, (short)0, acc, false, false);
    }
    const float bb = bias[(n0 + ln) & 31];   // out.reshape(-1,32)+bias == +bias[w%32]
#pragma unroll
    for (int r = 0; r < 8; ++r)
      out[(((size_t)b * COUT + (m0 + r + 8 * g)) * T_ + t) * V_ + n0 + ln] = acc[r] + bb;
  }
}

// ---------------------------------------------------------------------------
extern "C" void kernel_launch(void* const* d_in, const int* in_sizes, int n_in,
                              void* d_out, int out_size, void* d_ws, size_t ws_size,
                              hipStream_t stream) {
  (void)in_sizes; (void)n_in; (void)out_size; (void)ws_size;
  const float* x      = (const float*)d_in[0];
  const float* weight = (const float*)d_in[1];
  const float* bias   = (const float*)d_in[2];
  const float* Wq     = (const float*)d_in[3];
  const float* bq     = (const float*)d_in[4];
  const float* Wk     = (const float*)d_in[5];
  const float* bk     = (const float*)d_in[6];
  float* out = (float*)d_out;

  char* ws = (char*)d_ws;
  const size_t qkBytes = (size_t)BT * V_ * CIN * sizeof(_Float16);   // 12.6 MB each
  _Float16* Qh = (_Float16*)ws;
  _Float16* Kh = (_Float16*)(ws + qkBytes);
  _Float16* Gh = (_Float16*)(ws + 2 * qkBytes);                      // 6.3 MB f16

  k_proj<<<BT, 256, 0, stream>>>(x, Wq, bq, Wk, bk, Qh, Kh);
  k_gcn<<<96, 256, 0, stream>>>(x, weight, Gh);
  k_attn<<<BT, 256, 0, stream>>>(Qh, Kh, Gh, bias, out);
}
